// NaiveSDPA_83511344103695
// MI455X (gfx1250) — matrix operand order repaired and checked by reading, not compiled, not run on hardware
//
#include <hip/hip_runtime.h>
#include <hip/hip_bf16.h>

typedef _Float16 h16 __attribute__((ext_vector_type(16)));
typedef _Float16 h8  __attribute__((ext_vector_type(8)));
typedef __fp16   h2p __attribute__((ext_vector_type(2)));   // cvt_pkrtz result type
typedef float    f8  __attribute__((ext_vector_type(8)));
typedef unsigned int u32x4 __attribute__((ext_vector_type(4)));
typedef int          i32x8 __attribute__((ext_vector_type(8)));
typedef int          i32x4 __attribute__((ext_vector_type(4)));

#define B_ 2
#define S_ 2048
#define H_ 16
#define D_ 128
#define QT (S_ / 16)

#if defined(__gfx1250__) && __has_builtin(__builtin_amdgcn_tensor_load_to_lds) && \
    __has_builtin(__builtin_amdgcn_s_wait_tensorcnt)
#define USE_TDM 1
#else
#define USE_TDM 0
#endif

constexpr float SCALE   = 0.08838834764831845f; // 1/sqrt(128)
constexpr float CAP     = 20.0f;
constexpr float INV_CAP = 1.0f / 20.0f;

union H16 {
    h16 v;
    h2p p[8];
    struct { h8 lo, hi; } s;
    _Float16 e[16];
};

// CDNA5 hardware tanh (32-bit trans op). v_nop satisfies the multicycle
// TRANS-op result hazard since the compiler can't schedule around inline asm.
__device__ __forceinline__ float tanh_hw(float x) {
#if defined(__gfx1250__)
    float r;
    asm volatile("v_tanh_f32 %0, %1\n\tv_nop" : "=v"(r) : "v"(x));
    return r;
#else
    return tanhf(x);
#endif
}

// convert two contiguous 8-float chunks into one f16 A/B fragment register set
__device__ __forceinline__ h16 cvt_8_8(const float* lo, const float* hi) {
    const float4* a = (const float4*)lo;
    const float4* b = (const float4*)hi;
    H16 u;
    float4 f;
    f = a[0]; u.p[0] = __builtin_amdgcn_cvt_pkrtz(f.x, f.y); u.p[1] = __builtin_amdgcn_cvt_pkrtz(f.z, f.w);
    f = a[1]; u.p[2] = __builtin_amdgcn_cvt_pkrtz(f.x, f.y); u.p[3] = __builtin_amdgcn_cvt_pkrtz(f.z, f.w);
    f = b[0]; u.p[4] = __builtin_amdgcn_cvt_pkrtz(f.x, f.y); u.p[5] = __builtin_amdgcn_cvt_pkrtz(f.z, f.w);
    f = b[1]; u.p[6] = __builtin_amdgcn_cvt_pkrtz(f.x, f.y); u.p[7] = __builtin_amdgcn_cvt_pkrtz(f.z, f.w);
    return u.v;
}

__device__ __forceinline__ float rowmax16(float v) {
#pragma unroll
    for (int mk = 8; mk >= 1; mk >>= 1) v = fmaxf(v, __shfl_xor(v, mk, 32));
    return v;
}
__device__ __forceinline__ float rowsum16(float v) {
#pragma unroll
    for (int mk = 8; mk >= 1; mk >>= 1) v += __shfl_xor(v, mk, 32);
    return v;
}

#if USE_TDM
// Issue one TDM 2-D tile load: 32 rows x 128 f32, row stride H_*D_ floats.
// D# layout per CDNA5 ISA ch.8 (group0: count/lds/global/type, group1: dims).
// 6-arg builtin form: (uint32x4, int32x8, int32x4, int32x4, int32x8, i32 cpol)
__device__ __forceinline__ void tdm_load_tile(const float* gsrc, unsigned ldsOff) {
    unsigned long long ga = (unsigned long long)(uintptr_t)gsrc;
    u32x4 g0;
    g0[0] = 1u;                                   // count=1, user descriptor
    g0[1] = ldsOff;                               // lds_addr (bytes)
    g0[2] = (unsigned)(ga & 0xFFFFFFFFu);         // global_addr[31:0]
    g0[3] = (unsigned)((ga >> 32) & 0x01FFFFFFu)  // global_addr[56:32]
            | (2u << 30);                         // type=2 ("image")
    i32x8 g1;
    g1[0] = (int)0x00020000u;          // data_size=2 (4 bytes/elem)
    g1[1] = (int)(128u << 16);         // tensor_dim0 = 128
    g1[2] = 0;                         // tensor_dim0 hi / tensor_dim1 lo
    g1[3] = (int)((128u << 16) | 1u);  // tensor_dim1 = 65536 rows | tile_dim0=128
    g1[4] = 32;                        // tile_dim1 = 32, tile_dim2 = 0
    g1[5] = (int)(H_ * D_);            // tensor_dim0_stride = 2048 elems
    g1[6] = 0;
    g1[7] = 0;
    i32x4 gz4  = {0, 0, 0, 0};                    // groups 2/3 unused (2-D tile)
    i32x8 gz8  = {0, 0, 0, 0, 0, 0, 0, 0};
    __builtin_amdgcn_tensor_load_to_lds(g0, g1, gz4, gz4, gz8, 0);
}
#endif

__global__ __launch_bounds__(32) void sdpa_fa_kernel(
    const float* __restrict__ Q, const float* __restrict__ K,
    const float* __restrict__ V, float* __restrict__ O) {
#if USE_TDM
    __shared__ __align__(16) float Ksf[2][32 * D_];   // K tile f32, double buffered
    __shared__ __align__(16) float Vsf[2][32 * D_];   // V tile f32, double buffered
#endif
    // packed-transposed V tile: word (feat*16 + kp) = f16 pair {key 2kp, 2kp+1}
    __shared__ __align__(16) unsigned VsT[D_ * 16];
    __shared__ __align__(16) _Float16 Ps[16 * 32];    // P tile row-major f16

    const int lane = threadIdx.x;
    const int half = lane >> 4;
    const int ln   = lane & 15;

    const int qt = blockIdx.x % QT;
    const int bh = blockIdx.x / QT;
    const int b  = bh / H_;
    const int h  = bh % H_;
    const int q0 = qt * 16;

    const size_t rs_ = (size_t)H_ * D_;
    const float* Qb = Q + ((size_t)b * S_ * H_ + h) * D_;
    const float* Kb = K + ((size_t)b * S_ * H_ + h) * D_;
    const float* Vb = V + ((size_t)b * S_ * H_ + h) * D_;
    float*       Ob = O + ((size_t)b * S_ * H_ + h) * D_;

    const int nkv = (q0 + 16 + 31) / 32;

#if USE_TDM
    const unsigned ldsK[2] = {(unsigned)(uintptr_t)&Ksf[0][0], (unsigned)(uintptr_t)&Ksf[1][0]};
    const unsigned ldsV[2] = {(unsigned)(uintptr_t)&Vsf[0][0], (unsigned)(uintptr_t)&Vsf[1][0]};
    // preload tile 0 asynchronously
    tdm_load_tile(Kb, ldsK[0]);
    tdm_load_tile(Vb, ldsV[0]);
#endif

    // ---- Q A-fragments: 4 chunks of 16x32 f16 (A layout: lane base = half*16) ----
    h16 qa[4];
    {
        const float* qrow = Qb + (size_t)(q0 + ln) * rs_;
#pragma unroll
        for (int kb = 0; kb < 4; ++kb) {
            const int c0 = kb * 32 + half * 16;
            qa[kb] = cvt_8_8(qrow + c0, qrow + c0 + 8);
        }
    }

    f8 o[8];
#pragma unroll
    for (int fb = 0; fb < 8; ++fb) o[fb] = {};
    float m[8], l[8];
#pragma unroll
    for (int r = 0; r < 8; ++r) { m[r] = -INFINITY; l[r] = 0.0f; }

    for (int kt = 0; kt < nkv; ++kt) {
        const int kv0 = kt * 32;
#if USE_TDM
        const int cur = kt & 1;
        if (kt + 1 < nkv) {
            // kick off next tile while this one computes; wait only for current
            tdm_load_tile(Kb + (size_t)(kv0 + 32) * rs_, ldsK[cur ^ 1]);
            tdm_load_tile(Vb + (size_t)(kv0 + 32) * rs_, ldsV[cur ^ 1]);
            __builtin_amdgcn_s_wait_tensorcnt(2);
        } else {
            __builtin_amdgcn_s_wait_tensorcnt(0);
        }
        __syncthreads();   // compiler ordering barrier for LDS reads below
#else
        if (kt + 1 < nkv) {
            __builtin_prefetch(Kb + (size_t)(kv0 + 32 + lane) * rs_, 0, 0);
            __builtin_prefetch(Vb + (size_t)(kv0 + 32 + lane) * rs_, 0, 0);
        }
#endif

        // ---- repack V tile into packed-transposed f16 LDS layout ----
        // lane owns key pair (2*ln, 2*ln+1) and feature half `half`
        {
#if USE_TDM
            const float* r0 = &Vsf[cur][(size_t)(2 * ln) * D_ + half * 64];
            const float* r1 = r0 + D_;
#else
            const float* r0 = Vb + (size_t)(kv0 + 2 * ln) * rs_ + half * 64;
            const float* r1 = r0 + rs_;
#endif
            unsigned* dst = &VsT[(half * 64) * 16 + ln];
#pragma unroll 16
            for (int e = 0; e < 64; ++e) {
                h2p pk = __builtin_amdgcn_cvt_pkrtz(r0[e], r1[e]);
                dst[e * 16] = __builtin_bit_cast(unsigned int, pk);
            }
        }

        // ---- scores S[16q x 32k]: two 16x16 WMMA blocks x 4 K-chunks ----
        f8 sc0 = {}, sc1 = {};
#pragma unroll
        for (int kb = 0; kb < 4; ++kb) {
            const int c0 = kb * 32 + half * 16;
#if USE_TDM
            const float* k0 = &Ksf[cur][(size_t)ln * D_ + c0];
            const float* k1 = &Ksf[cur][(size_t)(16 + ln) * D_ + c0];
#else
            const float* k0 = Kb + (size_t)(kv0 + ln) * rs_ + c0;
            const float* k1 = Kb + (size_t)(kv0 + 16 + ln) * rs_ + c0;
#endif
            h16 bf0 = cvt_8_8(k0, k0 + 8);
            sc0 = __builtin_amdgcn_wmma_f32_16x16x32_f16(
                false, qa[kb], false, bf0, (short)0, sc0, false, false);
            h16 bf1 = cvt_8_8(k1, k1 + 8);
            sc1 = __builtin_amdgcn_wmma_f32_16x16x32_f16(
                false, qa[kb], false, bf1, (short)0, sc1, false, false);
        }

        // ---- scale, tanh soft-cap (hw v_tanh_f32), causal mask, online softmax ----
        const bool diagTile = (kv0 + 31 > q0);
        float p0[8], p1[8];
#pragma unroll
        for (int r = 0; r < 8; ++r) {
            const int qi = q0 + r + half * 8;
            float s0 = CAP * tanh_hw(sc0[r] * (SCALE * INV_CAP));
            float s1 = CAP * tanh_hw(sc1[r] * (SCALE * INV_CAP));
            if (diagTile) {
                if (kv0 + ln > qi)      s0 = -INFINITY;
                if (kv0 + 16 + ln > qi) s1 = -INFINITY;
            }
            const float rm = rowmax16(fmaxf(s0, s1));
            const float mn = fmaxf(m[r], rm);
            const float e0 = __expf(s0 - mn);
            const float e1 = __expf(s1 - mn);
            const float sum  = rowsum16(e0 + e1);
            const float corr = __expf(m[r] - mn);
            l[r] = l[r] * corr + sum;
            m[r] = mn;
#pragma unroll
            for (int fb = 0; fb < 8; ++fb) o[fb][r] *= corr;
            p0[r] = e0;
            p1[r] = e1;
        }

        // ---- stage P (C layout -> LDS row-major 16x32 f16) ----
#pragma unroll
        for (int r = 0; r < 8; ++r) {
            const int row = r + half * 8;
            Ps[row * 32 + ln]      = (_Float16)p0[r];
            Ps[row * 32 + 16 + ln] = (_Float16)p1[r];
        }
        __syncthreads();

        // P as A-fragment: two contiguous 16B chunks per lane
        H16 pu;
        pu.s.lo = *(const h8*)&Ps[ln * 32 + half * 16];
        pu.s.hi = *(const h8*)&Ps[ln * 32 + half * 16 + 8];

        // ---- PV: 8 feature blocks; V B-fragment = 2x contiguous ds_load_b128 ----
#pragma unroll
        for (int fb = 0; fb < 8; ++fb) {
            const unsigned* vp = &VsT[(fb * 16 + ln) * 16 + half * 8];
            H16 vu;
            vu.s.lo = *(const h8*)vp;
            vu.s.hi = *(const h8*)(vp + 4);
            o[fb] = __builtin_amdgcn_wmma_f32_16x16x32_f16(
                false, pu.v, false, vu.v, (short)0, o[fb], false, false);
        }
        __syncthreads();   // protect VsT/Ps before next iteration overwrites
    }

    // ---- epilogue: divide by row sum, coalesced fp32 stores ----
#pragma unroll
    for (int r = 0; r < 8; ++r) {
        const float inv = 1.0f / l[r];
        float* orow = Ob + (size_t)(q0 + r + half * 8) * rs_;
#pragma unroll
        for (int fb = 0; fb < 8; ++fb) orow[fb * 16 + ln] = o[fb][r] * inv;
    }
}

extern "C" void kernel_launch(void* const* d_in, const int* in_sizes, int n_in,
                              void* d_out, int out_size, void* d_ws, size_t ws_size,
                              hipStream_t stream) {
    const float* Q = (const float*)d_in[0];
    const float* K = (const float*)d_in[1];
    const float* V = (const float*)d_in[2];
    // d_in[3] is the causal additive mask; implemented analytically in-kernel.
    float* O = (float*)d_out;
    (void)in_sizes; (void)n_in; (void)out_size; (void)d_ws; (void)ws_size;

    dim3 grid(B_ * H_ * QT);   // 4096 query tiles of 16 rows
    sdpa_fa_kernel<<<grid, 32, 0, stream>>>(Q, K, V, O);
}
